// MotionCompensationBlock_80315888435595
// MI455X (gfx1250) — matrix operand order repaired and checked by reading, not compiled, not run on hardware
//
#include <hip/hip_runtime.h>
#include <math.h>

typedef __attribute__((ext_vector_type(16))) __bf16 v16bf;
typedef __attribute__((ext_vector_type(8)))  float  v8f;

__device__ __forceinline__ __bf16 f2bf(float f) { return (__bf16)f; }

__device__ __forceinline__ v8f wmma_bf16(v16bf a, v16bf b, v8f c) {
  // D = A(16x32) * B(32x16) + C, f32 accumulate
  return __builtin_amdgcn_wmma_f32_16x16x32_bf16(false, a, false, b, (short)0, c,
                                                 false, false);
}

// ---------------------------------------------------------------------------
// Pack f32 OIHW conv weights (Cout x Cin*9) into WMMA A-fragment layout with
// tap-major K reindexing:  k' = tap*Cpad + ci  (Cpad = ceil(Cin/32)*32).
// Layout: [tm][kc][lane][16 bf16], zero-padded on M tail and ci >= Cin.
// ---------------------------------------------------------------------------
__global__ void pack_weights_kernel(const float* __restrict__ w, __bf16* __restrict__ dst,
                                    int Cout, int Cin, int Cpad, int Kchunks, int total) {
  int i = blockIdx.x * blockDim.x + threadIdx.x;
  if (i >= total) return;
  int e = i & 15, lane = (i >> 4) & 31;
  int chunk = i >> 9;                       // tm*Kchunks + kc
  int tm = chunk / Kchunks, kc = chunk - tm * Kchunks;
  int m = tm * 16 + (lane & 15);
  int hi = lane >> 4;
  int kp = kc * 32 + e + (hi << 4);   // A-fragment: contiguous k = 16*hi + e
  int tap = kp / Cpad;
  int ci  = kp - tap * Cpad;
  float v = (m < Cout && ci < Cin) ? w[(long)m * (Cin * 9) + ci * 9 + tap] : 0.f;
  dst[i] = f2bf(v);
}

// ---------------------------------------------------------------------------
// Generic 3x3 conv (stride 1, pad 1, NCHW) as implicit GEMM, tap-major K.
// H/W compile-time -> fragment gathers are one base address + 16 loads with
// immediate offsets, clause-batched. Zero-padding done by MULTIPLYING by a
// float mask (not select) so the compiler cannot predicate the loads.
// ---------------------------------------------------------------------------
template <int H, int W, bool PAD>
__global__ __launch_bounds__(256) void conv3x3_wmma_kernel(
    const float* __restrict__ in, const __bf16* __restrict__ wpk,
    const float* __restrict__ bias, float* __restrict__ out,
    int Cin, int Cout, int Cc /* = Cpad/32 */, int tiles_m, int tiles_n)
{
  constexpr int HW = H * W;
  const int lane = threadIdx.x & 31;
  const int wid  = (blockIdx.x * blockDim.x + threadIdx.x) >> 5;
  if (wid >= tiles_m * tiles_n) return;          // wave-uniform exit
  const int tm = wid / tiles_n, tn = wid - tm * tiles_n;
  const int Kch = 9 * Cc;
  const int lm = lane & 15, hi = lane >> 4;
  const int p0 = tn * 64;
  const int b0 = p0 / HW;                        // wave-uniform (64 | HW)
  const int r0 = p0 - b0 * HW;
  const float* inb = in + (long)b0 * Cin * HW;
  float* outb = out + (long)b0 * Cout * HW;
  const v16bf* apk = (const v16bf*)wpk + (long)tm * Kch * 32 + lane;

  int ph[4], pw[4];
#pragma unroll
  for (int s = 0; s < 4; ++s) {
    int r = r0 + s * 16 + lm;
    ph[s] = r / W; pw[s] = r - ph[s] * W;
  }

  v8f acc[4] = {};
  int kc = 0;
  for (int tap = 0; tap < 9; ++tap) {
    const int ty = tap / 3 - 1, tx = tap - (tap / 3) * 3 - 1;
    float pokf[4]; int poff[4];
#pragma unroll
    for (int s = 0; s < 4; ++s) {
      int y = ph[s] + ty, xx = pw[s] + tx;
      bool ok = ((unsigned)y < (unsigned)H) && ((unsigned)xx < (unsigned)W);
      pokf[s] = ok ? 1.f : 0.f;
      poff[s] = ok ? (y * W + xx) : 0;
    }
    for (int kcc = 0; kcc < Cc; ++kcc, ++kc) {
      __builtin_prefetch((const void*)(apk + (kc + 1) * 32), 0, 1);
      v16bf a = apk[kc * 32];                    // 2x global_load_b128 (clause)
      const int ciFrag = kcc * 32 + (hi << 4);
      v16bf frag[4];
#pragma unroll
      for (int s = 0; s < 4; ++s) {
        const float* ps = inb + ciFrag * HW + poff[s];   // one address per (s)
#pragma unroll
        for (int e = 0; e < 16; ++e) {
          float v = ps[e * HW];                  // immediate-offset load
          float msk = PAD ? ((ciFrag + e < Cin) ? pokf[s] : 0.f) : pokf[s];
          frag[s][e] = f2bf(v * msk);            // mask by multiply, no branch
        }
      }
#pragma unroll
      for (int s = 0; s < 4; ++s) acc[s] = wmma_bf16(a, frag[s], acc[s]);
    }
  }
#pragma unroll
  for (int s = 0; s < 4; ++s) {
    int r = r0 + s * 16 + lm;
    float* po = outb + (tm * 16 + (hi << 3)) * HW + r;
#pragma unroll
    for (int t = 0; t < 8; ++t) {
      int mo = tm * 16 + t + (hi << 3);
      if (mo < Cout)
        po[t * HW] = acc[s][t] + bias[mo];
    }
  }
}

// ---------------------------------------------------------------------------
// Modulated deformable conv (DCNv2, 8 groups), tap-major K, templated on
// CG = C/8 and H (=W). Bilinear setup hoisted per group (1-2 per fragment);
// corner gathers use 4 base addresses + immediate channel offsets; masking
// is folded into the 4 corner weights (pure multiplies, branchless).
// ---------------------------------------------------------------------------
template <int CG, int H>
__global__ __launch_bounds__(256) void dcn_wmma_kernel(
    const float* __restrict__ x,    // (B,2,C,H,W), frame 1 is sampled
    const float* __restrict__ om,   // (B,216,H,W): dy | dx | sigmoid(mask)
    const __bf16* __restrict__ wpk, const float* __restrict__ bias,
    float* __restrict__ out, int tiles_n)
{
  constexpr int W   = H;
  constexpr int HW  = H * W;
  constexpr int C   = CG * 8;
  constexpr int Cc  = C / 32;
  constexpr int Kch = 9 * Cc;
  constexpr int TM  = C / 16;
  constexpr int SUB = (CG < 16) ? CG : 16;   // channels per group inside a fragment
  constexpr int NG  = 16 / SUB;              // groups per fragment
  const int lane = threadIdx.x & 31;
  const int wid  = (blockIdx.x * blockDim.x + threadIdx.x) >> 5;
  if (wid >= TM * tiles_n) return;
  const int tm = wid / tiles_n, tn = wid - tm * tiles_n;
  const int lm = lane & 15, hi = lane >> 4;
  const int p0 = tn * 64;
  const int b0 = p0 / HW;                    // wave-uniform
  const int r0 = p0 - b0 * HW;
  const float* xb  = x  + (long)(b0 * 2 + 1) * C * HW;
  const float* omb = om + (long)b0 * 216 * HW;
  float* outb = out + (long)b0 * C * HW;
  const v16bf* apk = (const v16bf*)wpk + (long)tm * Kch * 32 + lane;

  int ph[4], pw[4], pix[4];
#pragma unroll
  for (int s = 0; s < 4; ++s) {
    int r = r0 + s * 16 + lm;
    ph[s] = r / W; pw[s] = r - ph[s] * W;
    pix[s] = r;                               // r == ph*W + pw
  }

  v8f acc[4] = {};
  int kc = 0;
  for (int tap = 0; tap < 9; ++tap) {
    const int ty = tap / 3 - 1, tx = tap - (tap / 3) * 3 - 1;
    for (int kcc = 0; kcc < Cc; ++kcc, ++kc) {
      __builtin_prefetch((const void*)(apk + (kc + 1) * 32), 0, 1);
      v16bf a = apk[kc * 32];
      const int ciFrag = kcc * 32 + (hi << 4);
      v16bf frag[4];
#pragma unroll
      for (int s = 0; s < 4; ++s) {
#pragma unroll
        for (int gi = 0; gi < NG; ++gi) {
          const int ci0 = ciFrag + gi * SUB;
          const int g   = ci0 / CG;            // shift (CG is power of two)
          const float* po = omb + (g * 9 + tap) * HW + pix[s];
          float dy = po[0];
          float dx = po[72 * HW];
          float mk = po[144 * HW];
          float py = (float)(ph[s] + ty) + dy;
          float px = (float)(pw[s] + tx) + dx;
          float y0f = floorf(py), x0f = floorf(px);
          int   y0 = (int)y0f,   x0 = (int)x0f;
          float fy = py - y0f,   fx = px - x0f;
          int y0c = min(max(y0, 0), H - 1), y1c = min(max(y0 + 1, 0), H - 1);
          int x0c = min(max(x0, 0), W - 1), x1c = min(max(x0 + 1, 0), W - 1);
          float vy0 = ((unsigned)y0       < (unsigned)H) ? (1.f - fy) : 0.f;
          float vy1 = ((unsigned)(y0 + 1) < (unsigned)H) ? fy : 0.f;
          float vx0 = ((unsigned)x0       < (unsigned)W) ? (1.f - fx) : 0.f;
          float vx1 = ((unsigned)(x0 + 1) < (unsigned)W) ? fx : 0.f;
          float w00 = vy0 * vx0 * mk, w01 = vy0 * vx1 * mk;
          float w10 = vy1 * vx0 * mk, w11 = vy1 * vx1 * mk;
          const float* pc = xb + ci0 * HW;     // channel base for this group
          const float* p00 = pc + y0c * W + x0c;
          const float* p01 = pc + y0c * W + x1c;
          const float* p10 = pc + y1c * W + x0c;
          const float* p11 = pc + y1c * W + x1c;
#pragma unroll
          for (int e2 = 0; e2 < SUB; ++e2) {   // immediate-offset corner loads
            float v = w00 * p00[e2 * HW] + w01 * p01[e2 * HW]
                    + w10 * p10[e2 * HW] + w11 * p11[e2 * HW];
            frag[s][gi * SUB + e2] = f2bf(v);
          }
        }
      }
#pragma unroll
      for (int s = 0; s < 4; ++s) acc[s] = wmma_bf16(a, frag[s], acc[s]);
    }
  }
#pragma unroll
  for (int s = 0; s < 4; ++s) {
    int r = r0 + s * 16 + lm;
    float* po = outb + (tm * 16 + (hi << 3)) * HW + r;
#pragma unroll
    for (int t = 0; t < 8; ++t) {
      int mo = tm * 16 + t + (hi << 3);
      po[t * HW] = acc[s][t] + bias[mo];
    }
  }
}

// ---------------------------------------------------------------------------
// Elementwise helpers
// ---------------------------------------------------------------------------
__global__ void sigmoid_mask_kernel(float* om, int HW, int total) { // total = B*72*HW
  int i = blockIdx.x * blockDim.x + threadIdx.x;
  if (i >= total) return;
  int b = i / (72 * HW); int r = i - b * 72 * HW;
  long idx = ((long)b * 216 + 144) * HW + r;
  float v = om[idx];
  om[idx] = 1.f / (1.f + __expf(-v));
}

__global__ void concat2_kernel(const float* __restrict__ a, const float* __restrict__ b,
                               float* __restrict__ dst, int C1, int C2, int HW, int total) {
  int i = blockIdx.x * blockDim.x + threadIdx.x;
  if (i >= total) return;
  int Ct = C1 + C2;
  int bb = i / (Ct * HW); int r = i - bb * Ct * HW;
  int c = r / HW; int p = r - c * HW;
  dst[i] = (c < C1) ? a[((long)bb * C1 + c) * HW + p]
                    : b[((long)bb * C2 + (c - C1)) * HW + p];
}

__global__ void upsample2_kernel(const float* __restrict__ in, float* __restrict__ out,
                                 int H, int W, int total) { // total = B*C*4*H*W
  int i = blockIdx.x * blockDim.x + threadIdx.x;
  if (i >= total) return;
  int W2 = 2 * W, H2 = 2 * H;
  int p = i;
  int ow = p % W2; p /= W2;
  int oh = p % H2; p /= H2;            // p = b*C + c
  float sh = fmaxf((oh + 0.5f) * 0.5f - 0.5f, 0.f);
  float sw = fmaxf((ow + 0.5f) * 0.5f - 0.5f, 0.f);
  int h0 = (int)sh; int h1 = min(h0 + 1, H - 1); float th = sh - (float)h0;
  int w0 = (int)sw; int w1 = min(w0 + 1, W - 1); float tw = sw - (float)w0;
  const float* src = in + (long)p * H * W;
  float v = (src[h0 * W + w0] * (1.f - th) + src[h1 * W + w0] * th) * (1.f - tw)
          + (src[h0 * W + w1] * (1.f - th) + src[h1 * W + w1] * th) * tw;
  out[i] = v;
}

// ---------------------------------------------------------------------------
// Host orchestration
// ---------------------------------------------------------------------------
extern "C" void kernel_launch(void* const* d_in, const int* in_sizes, int n_in,
                              void* d_out, int out_size, void* d_ws, size_t ws_size,
                              hipStream_t stream) {
  (void)in_sizes; (void)n_in; (void)out_size; (void)ws_size;
  const float* x[3]     = {(const float*)d_in[0],  (const float*)d_in[2],  (const float*)d_in[4]};
  const float* flow[3]  = {(const float*)d_in[1],  (const float*)d_in[3],  (const float*)d_in[5]};
  const float* off_w[3] = {(const float*)d_in[6],  (const float*)d_in[12], (const float*)d_in[18]};
  const float* off_b[3] = {(const float*)d_in[7],  (const float*)d_in[13], (const float*)d_in[19]};
  const float* co_w[3]  = {(const float*)d_in[8],  (const float*)d_in[14], (const float*)d_in[20]};
  const float* co_b[3]  = {(const float*)d_in[9],  (const float*)d_in[15], (const float*)d_in[21]};
  const float* dcn_w[3] = {(const float*)d_in[10], (const float*)d_in[16], (const float*)d_in[22]};
  const float* dcn_b[3] = {(const float*)d_in[11], (const float*)d_in[17], (const float*)d_in[23]};
  const float* ch_w[2]  = {(const float*)d_in[24], (const float*)d_in[28]};
  const float* ch_b[2]  = {(const float*)d_in[25], (const float*)d_in[29]};
  const float* ft_w[2]  = {(const float*)d_in[26], (const float*)d_in[30]};
  const float* ft_b[2]  = {(const float*)d_in[27], (const float*)d_in[31]};

  const int Cl[3] = {64, 128, 256};
  const int Hl[3] = {128, 64, 32};
  const int B = 4;

  // ---- workspace layout ----
  __bf16* wpk    = (__bf16*)d_ws;         // packed bf16 weights (4,091,904 used)
  float* fbase   = (float*)d_ws + 2100000;
  float* buf_cat = fbase;                 // off_in / concat(feat, low)   8,519,680
  float* buf_of  = buf_cat + 8519680;     // off_feat, reused as dcn feat 4,194,304
  float* buf_om  = buf_of + 4194304;      // om, reused as low           14,155,776
  float* buf_up  = buf_om + 14155776;     // upsampled coarser feature    8,388,608

  float* outp[3] = {(float*)d_out,
                    (float*)d_out + 4194304,
                    (float*)d_out + 4194304 + 2097152};

  // ---- pack all 13 conv weights (tap-major, padded-ci) ----
  const float* wsrc[13] = {off_w[0], co_w[0], dcn_w[0],
                           off_w[1], co_w[1], dcn_w[1],
                           off_w[2], co_w[2], dcn_w[2],
                           ch_w[0], ft_w[0], ch_w[1], ft_w[1]};
  const int wco[13] = {64, 216, 64, 128, 216, 128, 256, 216, 256, 64, 64, 128, 128};
  const int wci[13] = {130, 64, 64, 258, 128, 128, 514, 256, 256, 128, 128, 256, 256};
  long wpo[13];
  {
    long wo = 0;
    for (int i = 0; i < 13; ++i) {
      int Mt = (wco[i] + 15) / 16;
      int Cc = (wci[i] + 31) / 32;
      int Kch = 9 * Cc;
      int total = Mt * Kch * 512;
      wpo[i] = wo;
      pack_weights_kernel<<<(total + 255) / 256, 256, 0, stream>>>(
          wsrc[i], wpk + wo, wco[i], wci[i], Cc * 32, Kch, total);
      wo += total;
    }
  }

  auto launch_conv = [&](const float* in, int widx, const float* bias,
                         float* out, int Cin, int Cout, int H) {
    int tiles_m = (Cout + 15) / 16;
    int Cc = (Cin + 31) / 32;
    bool pad = (Cc * 32 != Cin);
    int tiles_n = (B * H * H) / 64;
    int blocks = (tiles_m * tiles_n + 7) / 8;
    const __bf16* wp = wpk + wpo[widx];
#define LAUNCH_CONV(HH)                                                        \
    do {                                                                       \
      if (pad)                                                                 \
        conv3x3_wmma_kernel<HH, HH, true><<<blocks, 256, 0, stream>>>(         \
            in, wp, bias, out, Cin, Cout, Cc, tiles_m, tiles_n);               \
      else                                                                     \
        conv3x3_wmma_kernel<HH, HH, false><<<blocks, 256, 0, stream>>>(        \
            in, wp, bias, out, Cin, Cout, Cc, tiles_m, tiles_n);               \
    } while (0)
    if (H == 128) LAUNCH_CONV(128);
    else if (H == 64) LAUNCH_CONV(64);
    else LAUNCH_CONV(32);
#undef LAUNCH_CONV
  };

  for (int lvl = 2; lvl >= 0; --lvl) {
    const int C = Cl[lvl], H = Hl[lvl], HW = H * H;
    // off_in = concat(x_l (B,2C,H,W view), flow_l (B,2,H,W))
    {
      int total = B * (2 * C + 2) * HW;
      concat2_kernel<<<(total + 255) / 256, 256, 0, stream>>>(
          x[lvl], flow[lvl], buf_cat, 2 * C, 2, HW, total);
    }
    launch_conv(buf_cat, lvl * 3 + 0, off_b[lvl], buf_of, 2 * C + 2, C, H);
    launch_conv(buf_of, lvl * 3 + 1, co_b[lvl], buf_om, C, 216, H);
    {
      int total = B * 72 * HW;
      sigmoid_mask_kernel<<<(total + 255) / 256, 256, 0, stream>>>(buf_om, HW, total);
    }
    float* dcn_out = (lvl == 2) ? outp[2] : buf_of;   // off_feat is dead now
    {
      int tiles_m = C / 16;
      int tiles_n = (B * HW) / 64;
      int blocks = (tiles_m * tiles_n + 7) / 8;
      const __bf16* wp = wpk + wpo[lvl * 3 + 2];
      if (C == 64)
        dcn_wmma_kernel<8, 128><<<blocks, 256, 0, stream>>>(
            x[lvl], buf_om, wp, dcn_b[lvl], dcn_out, tiles_n);
      else if (C == 128)
        dcn_wmma_kernel<16, 64><<<blocks, 256, 0, stream>>>(
            x[lvl], buf_om, wp, dcn_b[lvl], dcn_out, tiles_n);
      else
        dcn_wmma_kernel<32, 32><<<blocks, 256, 0, stream>>>(
            x[lvl], buf_om, wp, dcn_b[lvl], dcn_out, tiles_n);
    }
    float* feat = dcn_out;
    if (lvl < 2) {
      float* low = buf_om;                              // om is dead now
      launch_conv(buf_up, 9 + lvl * 2, ch_b[lvl], low, Cl[lvl + 1], C, H);
      int total = B * 2 * C * HW;
      concat2_kernel<<<(total + 255) / 256, 256, 0, stream>>>(
          dcn_out, low, buf_cat, C, C, HW, total);
      launch_conv(buf_cat, 10 + lvl * 2, ft_b[lvl], outp[lvl], 2 * C, C, H);
      feat = outp[lvl];
    }
    if (lvl > 0) {
      int total = B * C * 4 * HW;
      upsample2_kernel<<<(total + 255) / 256, 256, 0, stream>>>(feat, buf_up, H, H, total);
    }
  }
}
